// MultiHeadedCrossAttention_73263552135572
// MI455X (gfx1250) — matrix operand order repaired and checked by reading, not compiled, not run on hardware
//
#include <hip/hip_runtime.h>

// ---------------- CDNA5 WMMA bf16 plumbing ----------------
typedef __bf16 bf16;
typedef __attribute__((ext_vector_type(16))) __bf16 v16bf;
typedef __attribute__((ext_vector_type(8)))  float  v8f;
typedef unsigned int u32;
typedef __attribute__((ext_vector_type(4))) u32 v4u;
typedef __attribute__((ext_vector_type(8))) int v8i;
typedef __attribute__((ext_vector_type(4))) int v4i;

union Frag { v16bf v; uint4 q[2]; };

// A-matrix fragment (16x32 bf16, M x K), per ISA 7.12.2:
// lane half h: elements 0..7 -> K = kb+8h .. kb+8h+7 ; elements 8..15 -> K = kb+16+8h ..
__device__ __forceinline__ v16bf load_frag_a(const bf16* base, int K, int kb, int lane) {
    const int m = lane & 15, h = lane >> 4;
    const bf16* p = base + m * K + kb + 8 * h;
    Frag f;
    f.q[0] = *(const uint4*)(p);
    f.q[1] = *(const uint4*)(p + 16);
    return f.v;
}

// B-matrix fragment (32x16 bf16, K x N), from a [N][K] (transposed) weight array:
// lane = N, elements 0..7 -> K = kb+8h .. kb+8h+7 ; elements 8..15 -> K = kb+16+8h ..
__device__ __forceinline__ v16bf load_frag_b(const bf16* wt, int K, int kb, int n0, int lane) {
    const int n = lane & 15, h = lane >> 4;
    const bf16* p = wt + (size_t)(n0 + n) * K + kb + 8 * h;
    Frag f;
    f.q[0] = *(const uint4*)(p);
    f.q[1] = *(const uint4*)(p + 16);
    return f.v;
}

__device__ __forceinline__ v8f wmma_bf16(v16bf a, v16bf b, v8f c) {
    // (neg_a, A, neg_b, B, c_mod, C, reuse_a, reuse_b)
    return __builtin_amdgcn_wmma_f32_16x16x32_bf16(false, a, false, b, (short)0, c, false, false);
}

// ---------------- problem constants ----------------
#define BB   2
#define II   2048
#define JJ   16
#define NKK  2048
#define DM   256
#define DH   512
#define HH   8
#define DKH  64
#define IPB  4            // i-rows per fused workgroup (M-tile = 64)

// ---------------- prep: transpose + bf16 convert weights ----------------
__global__ void transpose_bf16_kernel(const float* __restrict__ W, bf16* __restrict__ WT,
                                      int K, int N) {
    int idx = blockIdx.x * 256 + threadIdx.x;
    if (idx >= K * N) return;
    int k = idx / N, n = idx - k * N;
    WT[(size_t)n * K + k] = (bf16)W[idx];
}

__global__ void transpose_f32_kernel(const float* __restrict__ W, float* __restrict__ WT,
                                     int K, int N) {
    int idx = blockIdx.x * 256 + threadIdx.x;
    if (idx >= K * N) return;
    int k = idx / N, n = idx - k * N;
    WT[(size_t)n * K + k] = W[idx];
}

// ---------------- projections: Y[M][N] = X[M][K] @ W  (WT = [N][K] bf16) ----------------
// one block = 16 rows, 256 threads = 8 waves, K <= 256
__global__ void proj_gemm_kernel(const float* __restrict__ X, const bf16* __restrict__ WT,
                                 float* __restrict__ Y, int N, int K) {
    __shared__ __align__(16) bf16 Xs[16 * 256];
    const int m0 = blockIdx.x * 16;
    const int tot = 16 * K;
    for (int idx = threadIdx.x; idx < tot; idx += 256)
        Xs[idx] = (bf16)X[(size_t)m0 * K + idx];
    __syncthreads();

    const int wave = threadIdx.x >> 5, lane = threadIdx.x & 31;
    const int ntiles = N >> 4;
    for (int tile = wave; tile < ntiles; tile += 8) {
        const int n0 = tile << 4;
        v8f acc = {};
        #pragma unroll 4
        for (int kb = 0; kb < K; kb += 32) {
            v16bf a = load_frag_a(Xs, K, kb, lane);
            v16bf w = load_frag_b(WT, K, kb, n0, lane);
            acc = wmma_bf16(a, w, acc);
        }
        const int n = lane & 15, hr = lane >> 4;
        #pragma unroll
        for (int r = 0; r < 8; ++r)
            Y[(size_t)(m0 + r + 8 * hr) * N + n0 + n] = acc[r];
    }
}

// ---------------- fused: pos-MLP + gather + attention + Wout ----------------
// one block per 4 consecutive i rows; 512 threads = 16 waves; 256KB dynamic LDS.
// Pos tile is DMA'd global->LDS by the Tensor Data Mover, then converted to bf16.
// Per wave: one B-fragment is reused by 4 row-block WMMAs (x4 less L2 weight traffic,
// 4 independent WMMAs in flight per load).
__global__ void __launch_bounds__(512)
fused_attn_kernel(const float* __restrict__ pos, const int* __restrict__ lidx,
                  const float* __restrict__ b1, const float* __restrict__ b2,
                  const bf16* __restrict__ W1T, const bf16* __restrict__ W2T,
                  const float* __restrict__ WoutT,
                  const float* __restrict__ kvF, const float* __restrict__ qhF,
                  float* __restrict__ out) {
    extern __shared__ unsigned char smem[];
    bf16* Apos   = (bf16*)smem;                  // [64][512]  pos tile bf16,  64KB (reused)
    bf16* hidden = (bf16*)(smem + 65536);        // [64][512]  relu(..@W1+b1), 64KB
    bf16* poskv  = (bf16*)(smem + 131072);       // [64][1024] ..@W2+b2,      128KB
    // phase-4 scratch aliases region 0 (Apos dead after layer 1):
    float* scores = (float*)smem;                // [4][16][8]   2KB
    float* probs  = (float*)(smem + 2048);       // [4][16][8]   2KB
    int*   idxs   = (int*)(smem + 4096);         // [4][16]
    float* qsh    = (float*)(smem + 8192);       // [4][512]     8KB
    float* agg    = (float*)(smem + 16384);      // [4][512]     8KB

    const int i0 = blockIdx.x * IPB;             // global row (b*II + i), IPB rows per block
    const int b  = i0 >> 11;                     // II == 2048, IPB divides 2048 -> same b
    const int wave = threadIdx.x >> 5, lane = threadIdx.x & 31;

    const float* prow = pos + (size_t)i0 * JJ * DH;   // 64 x 512 f32, contiguous

#if __has_builtin(__builtin_amdgcn_tensor_load_to_lds)
    // ---- TDM: DMA raw f32 pos tile (64x512, 128KB) into the poskv region (dead now).
    // D# per CDNA5 ISA 8.3-8.6: group0 {count=1, lds_addr, global_addr, type=2},
    // group1 {data_size=4B, tensor_dim0=512, dim0_stride=512, tensor_dim1=64, tile 512x64}.
    // Groups 2/3 + extended group are zero (2D tile, no iterate/gather/multicast).
    if (wave == 0) {
        const unsigned long long ga = (unsigned long long)prow;
        const v4u g0 = { 1u,                                   // count=1 (valid user desc, load)
                         131072u,                              // lds_addr = poskv region offset
                         (u32)(ga & 0xffffffffu),              // global_addr[31:0]
                         (u32)((ga >> 32) & 0x01ffffffu) | (2u << 30) };  // addr[56:32], type=2
        const v8i g1 = { (int)(2u << 16),          // data_size = 2 (4 bytes); no multicast/pad
                         (int)(512u << 16),        // tensor_dim0[15:0] = 512 (bits 63:48)
                         (int)(64u << 16),         // tensor_dim1[15:0] = 64  (bits 95:80)
                         (int)(512u << 16),        // tile_dim0 = 512         (bits 127:112)
                         64,                       // tile_dim1 = 64          (bits 143:128)
                         512,                      // tensor_dim0_stride = 512 (bits 191:160)
                         0, 0 };
        const v4i gz4 = { 0, 0, 0, 0 };
        const v8i gz8 = { 0, 0, 0, 0, 0, 0, 0, 0 };
        __builtin_amdgcn_tensor_load_to_lds(g0, g1, gz4, gz4, gz8, 0);
        __builtin_amdgcn_s_wait_tensorcnt(0);
    }
    __syncthreads();
    // ---- convert staged f32 -> bf16 A-tile
    const float* staged = (const float*)(smem + 131072);
    for (int idx = threadIdx.x; idx < IPB * JJ * DH; idx += 512)
        Apos[idx] = (bf16)staged[idx];
#else
    for (int idx = threadIdx.x; idx < IPB * JJ * DH; idx += 512)
        Apos[idx] = (bf16)prow[idx];
#endif
    __syncthreads();

    // ---- layer 1: hidden = relu(Apos @ W1 + b1)   (64x512, K=512)
    for (int tile = wave; tile < 32; tile += 16) {
        const int n0 = tile << 4;
        v8f acc[IPB] = {{}, {}, {}, {}};
        #pragma unroll 2
        for (int kb = 0; kb < DH; kb += 32) {
            v16bf w = load_frag_b(W1T, DH, kb, n0, lane);
            #pragma unroll
            for (int rb = 0; rb < IPB; ++rb) {
                v16bf a = load_frag_a(Apos + rb * 16 * DH, DH, kb, lane);
                acc[rb] = wmma_bf16(a, w, acc[rb]);
            }
        }
        const int n = lane & 15, hr = lane >> 4;
        const float bias = b1[n0 + n];
        #pragma unroll
        for (int rb = 0; rb < IPB; ++rb) {
            #pragma unroll
            for (int r = 0; r < 8; ++r) {
                float v = acc[rb][r] + bias;
                hidden[(rb * 16 + r + 8 * hr) * DH + n0 + n] = (bf16)(v > 0.f ? v : 0.f);
            }
        }
    }
    __syncthreads();

    // ---- layer 2: poskv = hidden @ W2 + b2   (64x1024, K=512)
    for (int tile = wave; tile < 64; tile += 16) {
        const int n0 = tile << 4;
        v8f acc[IPB] = {{}, {}, {}, {}};
        #pragma unroll 2
        for (int kb = 0; kb < DH; kb += 32) {
            v16bf w = load_frag_b(W2T, DH, kb, n0, lane);
            #pragma unroll
            for (int rb = 0; rb < IPB; ++rb) {
                v16bf a = load_frag_a(hidden + rb * 16 * DH, DH, kb, lane);
                acc[rb] = wmma_bf16(a, w, acc[rb]);
            }
        }
        const int n = lane & 15, hr = lane >> 4;
        const float bias = b2[n0 + n];
        #pragma unroll
        for (int rb = 0; rb < IPB; ++rb) {
            #pragma unroll
            for (int r = 0; r < 8; ++r)
                poskv[(rb * 16 + r + 8 * hr) * (2 * DH) + n0 + n] = (bf16)(acc[rb][r] + bias);
        }
    }
    __syncthreads();

    // ---- attention operands into reused region 0 (contiguous loads)
    if (threadIdx.x < IPB * JJ) idxs[threadIdx.x] = lidx[(size_t)i0 * JJ + threadIdx.x];
    for (int o = threadIdx.x; o < IPB * DH; o += 512) qsh[o] = qhF[(size_t)i0 * DH + o];
    __syncthreads();

    // ---- scores[ii][j][h] = scale * <qh[ii,h], kk[idx,h] + pos_k[ii,j,h]>
    //      pos_kv reshape: pos_k at h*128+d, pos_v at h*128+64+d ; kk at h*64+d, vv at 512+h*64+d
    {
        const int t = threadIdx.x;               // 512 = 4*16*8 combos
        const int ii = t >> 7, j = (t >> 3) & 15, h = t & 7;
        const float* kr = kvF + (size_t)(b * NKK + idxs[ii * JJ + j]) * (2 * DH) + h * DKH;
        const bf16*  pk = poskv + (ii * JJ + j) * (2 * DH) + h * 128;
        const float* qr = qsh + ii * DH + h * DKH;
        float s = 0.f;
        #pragma unroll 8
        for (int d = 0; d < DKH; ++d) s += qr[d] * (kr[d] + (float)pk[d]);
        scores[t] = s * 0.125f;                  // D_K^-0.5 ; layout [ii*128 + j*8 + h]
    }
    __syncthreads();

    // ---- softmax over j per (ii, h)
    {
        const int t = threadIdx.x;
        const int ii = t >> 7, h = t & 7;
        const float* sc = scores + ii * 128 + h;
        float mx = -1e30f;
        #pragma unroll
        for (int jj = 0; jj < JJ; ++jj) mx = fmaxf(mx, sc[jj * 8]);
        float ssum = 0.f;
        #pragma unroll
        for (int jj = 0; jj < JJ; ++jj) ssum += __expf(sc[jj * 8] - mx);
        probs[t] = __expf(scores[t] - mx) / ssum;
    }
    __syncthreads();

    // ---- agg[ii][h*64+d] = sum_j p[ii,j,h] * (vv[idx, h*64+d] + pos_v[ii,j, h*128+64+d])
    for (int o = threadIdx.x; o < IPB * DH; o += 512) {
        const int ii = o >> 9, hd = o & 511, h = hd >> 6, d = hd & 63;
        float a = 0.f;
        #pragma unroll
        for (int j = 0; j < JJ; ++j) {
            const float v = kvF[(size_t)(b * NKK + idxs[ii * JJ + j]) * (2 * DH) + DH + h * DKH + d]
                          + (float)poskv[(ii * JJ + j) * (2 * DH) + h * 128 + 64 + d];
            a += probs[ii * 128 + j * 8 + h] * v;
        }
        agg[o] = a;
    }
    __syncthreads();

    // ---- out[i0+ii, n] = agg[ii] . WoutT[n]   (f32 last layer)
    for (int o = threadIdx.x; o < IPB * DM; o += 512) {
        const int ii = o >> 8, n = o & 255;
        const float* wr = WoutT + (size_t)n * DH;
        const float* ar = agg + ii * DH;
        float s = 0.f;
        #pragma unroll 8
        for (int k2 = 0; k2 < DH; ++k2) s += ar[k2] * wr[k2];
        out[(size_t)i0 * DM + o] = s;            // (i0+ii)*256 + n == i0*256 + o
    }
}

// ---------------- host launch ----------------
extern "C" void kernel_launch(void* const* d_in, const int* in_sizes, int n_in,
                              void* d_out, int out_size, void* d_ws, size_t ws_size,
                              hipStream_t stream) {
    (void)in_sizes; (void)n_in; (void)out_size; (void)ws_size;
    const float* q    = (const float*)d_in[0];
    const float* k    = (const float*)d_in[1];
    const float* pos  = (const float*)d_in[2];
    const int*   lidx = (const int*)d_in[3];   // jnp.int64 downgrades to int32 without x64
    const float* Wq   = (const float*)d_in[4];
    const float* Wkv  = (const float*)d_in[5];
    const float* W1   = (const float*)d_in[6];
    const float* b1   = (const float*)d_in[7];
    const float* W2   = (const float*)d_in[8];
    const float* b2   = (const float*)d_in[9];
    const float* Wout = (const float*)d_in[10];
    float* out = (float*)d_out;

    char* ws = (char*)d_ws;
    size_t off = 0;
    bf16* WqT    = (bf16*)(ws + off);  off += (size_t)DH * DM * 2;          // [512][256]
    bf16* WkvT   = (bf16*)(ws + off);  off += (size_t)2 * DH * DM * 2;      // [1024][256]
    bf16* W1T    = (bf16*)(ws + off);  off += (size_t)DH * DH * 2;          // [512][512]
    bf16* W2T    = (bf16*)(ws + off);  off += (size_t)2 * DH * DH * 2;      // [1024][512]
    float* WoutT = (float*)(ws + off); off += (size_t)DM * DH * 4;          // [256][512]
    float* kvF   = (float*)(ws + off); off += (size_t)BB * NKK * 2 * DH * 4;// [4096][1024]
    float* qhF   = (float*)(ws + off); off += (size_t)BB * II * DH * 4;     // [4096][512]
    // total scratch ~28 MB

    transpose_bf16_kernel<<<(DM * DH + 255) / 256, 256, 0, stream>>>(Wq, WqT, DM, DH);
    transpose_bf16_kernel<<<(DM * 2 * DH + 255) / 256, 256, 0, stream>>>(Wkv, WkvT, DM, 2 * DH);
    transpose_bf16_kernel<<<(DH * DH + 255) / 256, 256, 0, stream>>>(W1, W1T, DH, DH);
    transpose_bf16_kernel<<<(DH * 2 * DH + 255) / 256, 256, 0, stream>>>(W2, W2T, DH, 2 * DH);
    transpose_f32_kernel<<<(DH * DM + 255) / 256, 256, 0, stream>>>(Wout, WoutT, DH, DM);

    // kv = k @ Wkv : M = B*NK = 4096 rows -> 256 blocks ; qh = q @ Wq
    proj_gemm_kernel<<<(BB * NKK) / 16, 256, 0, stream>>>(k, WkvT, kvF, 2 * DH, DM);
    proj_gemm_kernel<<<(BB * II) / 16, 256, 0, stream>>>(q, WqT, qhF, DH, DM);

    // fused pos-MLP + gather + attention + output projection
    // 64-row M-tile, 16 waves, 256KB dynamic LDS (CDNA5 allows 320KB/workgroup)
    fused_attn_kernel<<<(BB * II) / IPB, 512, 256 * 1024, stream>>>(
        pos, lidx, b1, b2, W1T, W2T, WoutT, kvF, qhF, out);
}
